// SingleHeadAttention_20675972563278
// MI455X (gfx1250) — compile-verified, matrix operand order checked by reading, not run on hardware
//
#include <hip/hip_runtime.h>
#include <hip/hip_bf16.h>

// ---------------------------------------------------------------------------
// Single-head causal attention, CDNA5 (gfx1250), wave32 + WMMA f16->f32.
// B=4, S=2048, E=A=1024.
// ---------------------------------------------------------------------------

#define BATCH 4
#define SEQ   2048
#define EMB   1024
#define ADIM  1024

typedef __attribute__((ext_vector_type(16))) _Float16 v16h;
typedef __attribute__((ext_vector_type(8)))  _Float16 v8h;
typedef __attribute__((ext_vector_type(8)))  float    v8f;

#if __has_builtin(__builtin_amdgcn_global_load_async_to_lds_b128) && \
    __has_builtin(__builtin_amdgcn_s_wait_asynccnt)
#define HAVE_ASYNC_LDS 1
// builtin expects pointers to "int __vector(4)" (per compiler diagnostic)
typedef __attribute__((__vector_size__(4 * sizeof(int)))) int async_v4i;
#else
#define HAVE_ASYNC_LDS 0
#endif

// A/B 16-bit fragment (16x32 / 32x16): per lane, two 16B chunks.
// lanes 0-15 hold K=[0..8)+[16..24), lanes 16-31 hold K=[8..16)+[24..32).
__device__ __forceinline__ v16h load_frag(const _Float16* __restrict__ p, int sel) {
    v8h lo = *(const v8h*)(p + (sel << 3));
    v8h hi = *(const v8h*)(p + 16 + (sel << 3));
    v16h r;
#pragma unroll
    for (int i = 0; i < 8; ++i) { r[i] = lo[i]; r[i + 8] = hi[i]; }
    return r;
}

__device__ __forceinline__ v8f wmma_f16(v16h a, v16h b, v8f c) {
    return __builtin_amdgcn_wmma_f32_16x16x32_f16(false, a, false, b, (short)0, c,
                                                  false, false);
}

// ---------------------------------------------------------------------------
// Stage 1: fp32 -> fp16 conversion
// ---------------------------------------------------------------------------
__global__ __launch_bounds__(256) void cvt_f32_f16_kernel(
    const float* __restrict__ src, _Float16* __restrict__ dst, int n) {
    int i = (blockIdx.x * 256 + threadIdx.x) * 4;
    if (i < n) {
        float4 v = *(const float4*)(src + i);
        dst[i + 0] = (_Float16)v.x;
        dst[i + 1] = (_Float16)v.y;
        dst[i + 2] = (_Float16)v.z;
        dst[i + 3] = (_Float16)v.w;
    }
}

// ---------------------------------------------------------------------------
// Stage 2: projections. Y = X @ W^T for W in {Wq,Wk,Wv}.
// Block = 256 threads (8 waves), output tile 128 rows x 128 cols.
// Wave = 32 rows x 64 cols = 8 WMMA C tiles (2 A frags x 4 B frags / k-step).
// Q,K stored [b*S + s, a] f16; V stored transposed Vt[b, a, s] f16.
// ---------------------------------------------------------------------------
__global__ __launch_bounds__(256) void proj_kernel(
    const _Float16* __restrict__ Xh, const _Float16* __restrict__ Wh,
    _Float16* __restrict__ Qh, _Float16* __restrict__ Kh,
    _Float16* __restrict__ Vt) {
    const int w    = blockIdx.z;                 // 0=Q, 1=K, 2=V
    const int m0   = blockIdx.x * 128;           // row block over B*S
    const int n0   = blockIdx.y * 128;           // col block over ADIM
    const int wave = threadIdx.x >> 5;
    const int lane = threadIdx.x & 31;
    const int sel  = lane >> 4;
    const int ln   = lane & 15;
    const int mrow = m0 + (wave >> 1) * 32;      // this wave's 32-row strip
    const int ncol = n0 + (wave & 1) * 64;       // this wave's 64 cols

    const _Float16* Wp  = Wh + (size_t)w * ADIM * EMB;
    const _Float16* Ar0 = Xh + (size_t)(mrow + ln) * EMB;
    const _Float16* Ar1 = Xh + (size_t)(mrow + 16 + ln) * EMB;

    v8f acc[2][4];
#pragma unroll
    for (int s2 = 0; s2 < 2; ++s2)
#pragma unroll
        for (int t = 0; t < 4; ++t) acc[s2][t] = (v8f)(0.0f);

    for (int k = 0; k < EMB; k += 32) {
        v16h a0 = load_frag(Ar0 + k, sel);
        v16h a1 = load_frag(Ar1 + k, sel);
#pragma unroll
        for (int t = 0; t < 4; ++t) {
            v16h b = load_frag(Wp + (size_t)(ncol + t * 16 + ln) * EMB + k, sel);
            acc[0][t] = wmma_f16(a0, b, acc[0][t]);
            acc[1][t] = wmma_f16(a1, b, acc[1][t]);
        }
    }

    if (w == 2) {
        // transposed store: Vt[b, a=ncol+t*16+n, s]; contiguous over vgpr r
        const int b = mrow / SEQ;
        const int s = mrow % SEQ;
#pragma unroll
        for (int s2 = 0; s2 < 2; ++s2)
#pragma unroll
            for (int t = 0; t < 4; ++t) {
                _Float16* dst = Vt + (size_t)(b * ADIM + ncol + t * 16 + ln) * SEQ +
                                s + s2 * 16 + sel * 8;
                v8h hv;
#pragma unroll
                for (int r = 0; r < 8; ++r) hv[r] = (_Float16)acc[s2][t][r];
                *(v8h*)dst = hv;
            }
    } else {
        _Float16* D = (w == 0) ? Qh : Kh;
#pragma unroll
        for (int s2 = 0; s2 < 2; ++s2)
#pragma unroll
            for (int t = 0; t < 4; ++t)
#pragma unroll
                for (int r = 0; r < 8; ++r) {
                    int m = mrow + s2 * 16 + r + sel * 8;
                    D[(size_t)m * ADIM + ncol + t * 16 + ln] = (_Float16)acc[s2][t][r];
                }
    }
}

// ---------------------------------------------------------------------------
// Stage 3: fused causal attention per (batch, 16-query tile).
// LDS: f32 score panel (128KB) + f16 prob panel (64KB) + Q panel (32KB) + red.
// ---------------------------------------------------------------------------
#define ATTN_LDS_BYTES (16 * SEQ * 4 + 16 * SEQ * 2 + 16 * ADIM * 2 + 256 * 4)

__global__ __launch_bounds__(256) void attn_kernel(
    const _Float16* __restrict__ Qh, const _Float16* __restrict__ Kh,
    const _Float16* __restrict__ Vt, float* __restrict__ out) {
    extern __shared__ char smem[];
    float*    Pnl = (float*)smem;                                    // [16][SEQ] f32
    _Float16* Ph  = (_Float16*)(smem + 16 * SEQ * 4);                // [16][SEQ] f16
    _Float16* Qp  = (_Float16*)(smem + 16 * SEQ * 4 + 16 * SEQ * 2); // [16][ADIM]
    float*    red = (float*)(smem + 16 * SEQ * 4 + 16 * SEQ * 2 + 16 * ADIM * 2);

    const int qb = blockIdx.x;            // query tile 0..S/16-1
    const int b  = blockIdx.y;            // batch
    const int q0 = qb * 16;
    const int nkt   = qb + 1;             // valid key tiles (causal)
    const int nkeys = q0 + 16;
    const int nk32  = (nkeys + 31) & ~31; // PV K-loop extent
    const int wave  = threadIdx.x >> 5;
    const int lane  = threadIdx.x & 31;
    const int sel   = lane >> 4;
    const int ln    = lane & 15;

    // zero f16 prob panel (padding columns must contribute 0 to PV)
    for (int i = threadIdx.x * 8; i < 16 * SEQ; i += 256 * 8)
        *(float4*)(Ph + i) = make_float4(0.f, 0.f, 0.f, 0.f);

    // stage Q panel (16 rows contiguous in Qh) -> LDS
    {
        const _Float16* src = Qh + (size_t)(b * SEQ + q0) * ADIM;
#if HAVE_ASYNC_LDS
        for (int i = threadIdx.x * 8; i < 16 * ADIM; i += 256 * 8) {
            __builtin_amdgcn_global_load_async_to_lds_b128(
                (async_v4i*)(src + i), (async_v4i*)(Qp + i), 0, 0);
        }
        __builtin_amdgcn_s_wait_asynccnt(0);
#else
        for (int i = threadIdx.x * 8; i < 16 * ADIM; i += 256 * 8)
            *(v8h*)(Qp + i) = *(const v8h*)(src + i);
#endif
    }
    __syncthreads();

    // ---- phase 1: scores = scale * Q Kt, causal masked, into f32 panel ----
    // Each wave handles PAIRS of key tiles sharing one A fragment.
    const float scale = 0.03125f;  // 1/sqrt(1024)
    for (int kt0 = wave * 2; kt0 < nkt; kt0 += 16) {
        const int  kt1  = kt0 + 1;
        const bool has1 = (kt1 < nkt);
        v8f acc0 = (v8f)(0.0f), acc1 = (v8f)(0.0f);
        const _Float16* Kr0 = Kh + (size_t)(b * SEQ + kt0 * 16 + ln) * ADIM;
        const _Float16* Kr1 =
            Kh + (size_t)(b * SEQ + (has1 ? kt1 : kt0) * 16 + ln) * ADIM;
        const _Float16* Arow = Qp + ln * ADIM;
        for (int k = 0; k < ADIM; k += 32) {
            v16h a  = load_frag(Arow + k, sel);  // from LDS
            v16h b0 = load_frag(Kr0 + k, sel);   // from global (L2-resident)
            v16h b1 = load_frag(Kr1 + k, sel);
            acc0 = wmma_f16(a, b0, acc0);
            acc1 = wmma_f16(a, b1, acc1);
        }
#pragma unroll
        for (int r = 0; r < 8; ++r) {
            int row = r + sel * 8;                // 0..15
            int col = kt0 * 16 + ln;
            float v = acc0[r] * scale;
            if (col > q0 + row) v = -3.0e38f;     // causal mask
            Pnl[row * SEQ + col] = v;
        }
        if (has1) {
#pragma unroll
            for (int r = 0; r < 8; ++r) {
                int row = r + sel * 8;
                int col = kt1 * 16 + ln;
                float v = acc1[r] * scale;
                if (col > q0 + row) v = -3.0e38f;
                Pnl[row * SEQ + col] = v;
            }
        }
    }
    __syncthreads();

    // ---- phase 2: row softmax; normalized probs written as f16 to Ph ----
    {
        const int row = threadIdx.x >> 4;
        const int t   = threadIdx.x & 15;
        const int nv  = q0 + row + 1;             // valid columns this row
        float* Pr = Pnl + row * SEQ;

        float mx = -3.0e38f;
        for (int c = t; c < nv; c += 16) mx = fmaxf(mx, Pr[c]);
        red[threadIdx.x] = mx;
        __syncthreads();
        if (t == 0) {
            float m = red[row * 16];
#pragma unroll
            for (int j = 1; j < 16; ++j) m = fmaxf(m, red[row * 16 + j]);
            red[row * 16] = m;
        }
        __syncthreads();
        mx = red[row * 16];
        __syncthreads();

        float sum = 0.f;
        for (int c = t; c < nv; c += 16) {
            float e = __expf(Pr[c] - mx);
            Pr[c] = e;
            sum += e;
        }
        red[threadIdx.x] = sum;
        __syncthreads();
        if (t == 0) {
            float s2 = red[row * 16];
#pragma unroll
            for (int j = 1; j < 16; ++j) s2 += red[row * 16 + j];
            red[row * 16] = s2;
        }
        __syncthreads();
        float inv = 1.0f / red[row * 16];
        _Float16* Phr = Ph + row * SEQ;
        for (int c = t; c < nv; c += 16) Phr[c] = (_Float16)(Pr[c] * inv);
    }
    __syncthreads();

    // ---- phase 3: O = P @ V. Wave handles 128 output cols = 8 C tiles ----
    const int a0 = wave * 128;
    v8f o[8];
#pragma unroll
    for (int t = 0; t < 8; ++t) o[t] = (v8f)(0.0f);

    for (int k = 0; k < nk32; k += 32) {
        v16h a = load_frag(Ph + ln * SEQ + k, sel);   // f16 probs from LDS
#pragma unroll
        for (int t = 0; t < 8; ++t) {
            v16h bb = load_frag(
                Vt + (size_t)(b * ADIM + a0 + t * 16 + ln) * SEQ + k, sel);
            o[t] = wmma_f16(a, bb, o[t]);
        }
    }

    // store with reference's round-to-4-decimals
#pragma unroll
    for (int t = 0; t < 8; ++t)
#pragma unroll
        for (int r = 0; r < 8; ++r) {
            int m   = r + sel * 8;
            int col = a0 + t * 16 + ln;
            float v = o[t][r];
            out[((size_t)(b * SEQ) + q0 + m) * ADIM + col] =
                rintf(v * 1.0e4f) * 1.0e-4f;
        }
}

// ---------------------------------------------------------------------------
// Host launcher
// ---------------------------------------------------------------------------
extern "C" void kernel_launch(void* const* d_in, const int* in_sizes, int n_in,
                              void* d_out, int out_size, void* d_ws, size_t ws_size,
                              hipStream_t stream) {
    const float* emb = (const float*)d_in[0];
    const float* Wk  = (const float*)d_in[1];
    const float* Wq  = (const float*)d_in[2];
    const float* Wv  = (const float*)d_in[3];
    float* out = (float*)d_out;

    // workspace layout (f16)
    char* ws = (char*)d_ws;
    _Float16* Xh = (_Float16*)ws;                       // [B*S, E]   16 MB
    ws += (size_t)BATCH * SEQ * EMB * 2;
    _Float16* Wh = (_Float16*)ws;                       // [3, A, E]   6 MB
    ws += (size_t)3 * ADIM * EMB * 2;
    _Float16* Qh = (_Float16*)ws;                       // [B*S, A]   16 MB
    ws += (size_t)BATCH * SEQ * ADIM * 2;
    _Float16* Kh = (_Float16*)ws;                       // [B*S, A]   16 MB
    ws += (size_t)BATCH * SEQ * ADIM * 2;
    _Float16* Vt = (_Float16*)ws;                       // [B, A, S]  16 MB

    // stage 1: convert to f16 (Wh slot order: 0=Wq, 1=Wk, 2=Wv)
    {
        int nx = BATCH * SEQ * EMB;
        cvt_f32_f16_kernel<<<(nx / 4 + 255) / 256, 256, 0, stream>>>(emb, Xh, nx);
        int nw = ADIM * EMB;
        cvt_f32_f16_kernel<<<(nw / 4 + 255) / 256, 256, 0, stream>>>(
            Wq, Wh + (size_t)0 * nw, nw);
        cvt_f32_f16_kernel<<<(nw / 4 + 255) / 256, 256, 0, stream>>>(
            Wk, Wh + (size_t)1 * nw, nw);
        cvt_f32_f16_kernel<<<(nw / 4 + 255) / 256, 256, 0, stream>>>(
            Wv, Wh + (size_t)2 * nw, nw);
    }

    // stage 2: projections (z: 0=Q, 1=K, 2=V)
    proj_kernel<<<dim3(BATCH * SEQ / 128, ADIM / 128, 3), 256, 0, stream>>>(
        Xh, Wh, Qh, Kh, Vt);

    // stage 3: fused causal attention
    attn_kernel<<<dim3(SEQ / 16, BATCH), 256, ATTN_LDS_BYTES, stream>>>(
        Qh, Kh, Vt, out);
}
